// ConvLSTMBlock_23596550324264
// MI455X (gfx1250) — matrix operand order repaired and checked by reading, not compiled, not run on hardware
//
#include <hip/hip_runtime.h>
#include <hip/hip_bf16.h>

#if defined(__has_include)
#if __has_include(<hip/amd_detail/amd_gfx1250_TDM.h>)
#define ATHENA_TDM_6ARG 1
#endif
#endif

typedef __attribute__((ext_vector_type(8)))  float          v8f;
typedef __attribute__((ext_vector_type(16))) __bf16         v16bf;
typedef __attribute__((ext_vector_type(8)))  unsigned short u16x8;
typedef __attribute__((ext_vector_type(4)))  unsigned int   u32x4;
typedef __attribute__((ext_vector_type(8)))  int            i32x8;
typedef __attribute__((ext_vector_type(4)))  int            i32x4;

#define BATCH   8
#define TSTEPS  16
#define CIN_X   32
#define HID     64
#define COUT    256         // 4*HID
#define HEIGHT  64
#define WIDTH   64
#define HW      4096
#define KTOT    864         // 96*9
#define KCHUNK  32
#define NCHUNKS 27          // 864/32
#define PITCHW  40          // ushorts per weight-tile LDS row (80B, 16B aligned, conflict-free)
#define PITCHI  72          // ushorts per im2col LDS row (144B)
#define EPSV    1e-3f
#define NEGSL   0.01f

// ---------------- bf16 split helpers (round-to-nearest-even) ----------------
static __device__ __forceinline__ unsigned short f32_to_bf16(float f) {
    unsigned u = __float_as_uint(f);
    unsigned r = (u + 0x7FFFu + ((u >> 16) & 1u)) >> 16;
    return (unsigned short)r;
}
static __device__ __forceinline__ float bf16_to_f32(unsigned short h) {
    return __uint_as_float(((unsigned)h) << 16);
}
// assemble a 16-element bf16 fragment from two contiguous 16B LDS runs
static __device__ __forceinline__ v16bf frag16(const unsigned short* p1,
                                               const unsigned short* p2) {
    union U { u16x8 h[2]; v16bf v; } u;
    u.h[0] = *(const u16x8*)p1;
    u.h[1] = *(const u16x8*)p2;
    return u.v;
}

// ---------------- TDM: DMA one 64x32 bf16 tile (pitch 80B via pad) ----------
#if __has_builtin(__builtin_amdgcn_tensor_load_to_lds)
#define ATHENA_HAVE_TDM 1
static __device__ __forceinline__ void tdm_load_tile(const unsigned short* gsrc,
                                                     unsigned lds_off) {
    unsigned long long ga = (unsigned long long)(uintptr_t)gsrc;
    u32x4 g0;
    g0.x = 1u;                                           // count=1 (valid D#)
    g0.y = lds_off;                                      // LDS byte address
    g0.z = (unsigned)(ga & 0xFFFFFFFFu);                 // global_addr[31:0]
    g0.w = (unsigned)((ga >> 32) & 0x1FFFFFFu)           // global_addr[56:32]
         | (2u << 30);                                   // type = 2 (image)
    i32x8 g1;
    // data_size=1 (2B), pad_enable, pad_interval=3 (16 DW = one 64B row),
    // pad_amount=3 (4 DW = 16B)  ->  LDS row pitch 80B == PITCHW*2
    g1[0] = (1 << 16) | (1 << 20) | (3 << 22) | (3 << 25);
    g1[1] = (int)((864u & 0xFFFFu) << 16);               // tensor_dim0[15:0]
    g1[2] = (int)((864u >> 16) | ((256u & 0xFFFFu) << 16)); // dim0 hi | dim1 lo
    g1[3] = (int)((256u >> 16) | (32u << 16));           // dim1 hi | tile_dim0=32
    g1[4] = (int)64;                                     // tile_dim1=64, tile_dim2=0
    g1[5] = (int)864;                                    // tensor_dim0_stride[31:0]
    g1[6] = 0;                                           // stride0 hi | stride1 lo
    g1[7] = 0;
    i32x4 z4 = {0, 0, 0, 0};
#ifdef ATHENA_TDM_6ARG
    i32x8 z8 = {0, 0, 0, 0, 0, 0, 0, 0};
    __builtin_amdgcn_tensor_load_to_lds(g0, g1, z4, z4, z8, 0);
#else
    __builtin_amdgcn_tensor_load_to_lds(g0, g1, z4, z4, 0);
#endif
}
#endif

// ---------------------------------------------------------------------------
// Zero h, c and stats; must run every launch (ws is not re-poisoned).
// ---------------------------------------------------------------------------
__global__ __launch_bounds__(256)
void init_kernel(float* __restrict__ h, float* __restrict__ c,
                 float* __restrict__ stats) {
    int idx = blockIdx.x * 256 + threadIdx.x;
    const int NSTATE = BATCH * HID * HW;
    if (idx < NSTATE) { h[idx] = 0.0f; c[idx] = 0.0f; }
    if (idx < 128)    stats[idx] = 0.0f;
}

// ---------------------------------------------------------------------------
// One-time: split fp32 conv weights into bf16 hi/lo matrices (256 x 864).
// ---------------------------------------------------------------------------
__global__ __launch_bounds__(256)
void prep_weights_kernel(const float* __restrict__ w,
                         unsigned short* __restrict__ whi,
                         unsigned short* __restrict__ wlo) {
    int i = blockIdx.x * 256 + threadIdx.x;
    if (i < COUT * KTOT) {
        float v = w[i];
        unsigned short h = f32_to_bf16(v);
        whi[i] = h;
        wlo[i] = f32_to_bf16(v - bf16_to_f32(h));
    }
}

// ---------------------------------------------------------------------------
// Implicit-GEMM 3x3 SAME conv, bf16x3 decomposition on V_WMMA_F32_16X16X32_BF16.
//   A (M=64 out-channels x K) = weights (hi/lo), staged to LDS by the TDM.
//   B (K x N=64 pixels)       = im2col of [x_t ; h], gathered + split inline.
// Block = 256 thr = 8 waves; wave (cw,pw) owns 16 ch x 32 px (2 accumulators).
// Per K-chunk of 32 a wave issues 6 WMMAs: hi*hi + hi*lo + lo*hi per subtile.
// ---------------------------------------------------------------------------
__global__ __launch_bounds__(256)
void conv_step_kernel(const float* __restrict__ x,      // (B,T,32,H,W)
                      const float* __restrict__ h,      // (B,64,H,W)
                      const unsigned short* __restrict__ whi,  // (256,864) bf16
                      const unsigned short* __restrict__ wlo,  // (256,864) bf16
                      const float* __restrict__ bias,   // (256)
                      float* __restrict__ cc,           // (B,256,H,W)
                      int tstep) {
    const int row = blockIdx.x;            // (b, y) image row
    const int b   = row >> 6;
    const int y   = row & 63;
    const int ch0 = blockIdx.y * 64;

    const int tid  = threadIdx.x;
    const int lane = tid & 31;
    const int wave = tid >> 5;
    const int cw   = wave & 3;             // channel sub-tile (16 ch)
    const int pw   = wave >> 2;            // pixel half (32 px)

    __shared__ __align__(16) unsigned short sWh[64 * PITCHW];
    __shared__ __align__(16) unsigned short sWl[64 * PITCHW];
    __shared__ __align__(16) unsigned short sIh[KCHUNK * PITCHI];
    __shared__ __align__(16) unsigned short sIl[KCHUNK * PITCHI];

    v8f acc0 = {};
    v8f acc1 = {};

    const int m   = lane & 15;
    const int r1  = (lane >> 4) * 8;       // K-run select for 16-bit A frag
    const int n0  = pw * 32;               // pixel base (ushort index)

    for (int kc = 0; kc < NCHUNKS; ++kc) {
        const int kbase = kc * KCHUNK;
        __syncthreads();                   // previous compute done -> LDS reusable

        // ---- weight tiles hi/lo: Tensor Data Mover (wave 0 only) ----
#ifdef ATHENA_HAVE_TDM
        if (tid < 32) {
            tdm_load_tile(whi + (ch0 * KTOT + kbase), (unsigned)(uintptr_t)sWh);
            tdm_load_tile(wlo + (ch0 * KTOT + kbase), (unsigned)(uintptr_t)sWl);
        }
#else
        {
            const int k  = tid & 31;
            const int c8 = tid >> 5;
#pragma unroll
            for (int p = 0; p < 8; ++p) {
                const int ch = p * 8 + c8;
                const int g  = (ch0 + ch) * KTOT + kbase + k;
                sWh[ch * PITCHW + k] = whi[g];
                sWl[ch * PITCHW + k] = wlo[g];
            }
        }
#endif
        // ---- im2col tile: gather + bf16 hi/lo split (all 256 threads) ----
        {
            const int xx = tid & 63;
            const int k4 = tid >> 6;
#pragma unroll
            for (int p = 0; p < 8; ++p) {
                const int kl  = p * 4 + k4;
                const int k   = kbase + kl;
                const int ci  = k / 9;
                const int tap = k - ci * 9;
                const int dy  = tap / 3 - 1;
                const int dx  = tap - (tap / 3) * 3 - 1;
                const int yy  = y + dy;
                const int xs  = xx + dx;
                float v = 0.0f;
                if ((unsigned)yy < (unsigned)HEIGHT && (unsigned)xs < (unsigned)WIDTH) {
                    if (ci < CIN_X)
                        v = x[((b * TSTEPS + tstep) * CIN_X + ci) * HW + yy * WIDTH + xs];
                    else
                        v = h[(b * HID + (ci - CIN_X)) * HW + yy * WIDTH + xs];
                }
                unsigned short hh = f32_to_bf16(v);
                sIh[kl * PITCHI + xx] = hh;
                sIl[kl * PITCHI + xx] = f32_to_bf16(v - bf16_to_f32(hh));
            }
        }
#ifdef ATHENA_HAVE_TDM
        if (tid < 32) __builtin_amdgcn_s_wait_tensorcnt(0);
#endif
        __syncthreads();

        // ---- 6 WMMAs over this K=32 chunk (bf16x3) ----
        {
            const unsigned short* wrh = &sWh[(cw * 16 + m) * PITCHW];
            const unsigned short* wrl = &sWl[(cw * 16 + m) * PITCHW];
            v16bf ah = frag16(wrh + r1, wrh + 16 + r1);
            v16bf al = frag16(wrl + r1, wrl + 16 + r1);

            v16bf bh0, bl0, bh1, bl1;
#pragma unroll
            for (int s = 0; s < 16; ++s) {
                const int kk = (s < 8) ? (r1 + s) : (16 + r1 + (s - 8));
                const int o  = kk * PITCHI + n0 + m;
                bh0[s] = __builtin_bit_cast(__bf16, sIh[o]);
                bl0[s] = __builtin_bit_cast(__bf16, sIl[o]);
                bh1[s] = __builtin_bit_cast(__bf16, sIh[o + 16]);
                bl1[s] = __builtin_bit_cast(__bf16, sIl[o + 16]);
            }

            acc0 = __builtin_amdgcn_wmma_f32_16x16x32_bf16(false, ah, false, bh0, (short)0, acc0, false, false);
            acc0 = __builtin_amdgcn_wmma_f32_16x16x32_bf16(false, ah, false, bl0, (short)0, acc0, false, false);
            acc0 = __builtin_amdgcn_wmma_f32_16x16x32_bf16(false, al, false, bh0, (short)0, acc0, false, false);
            acc1 = __builtin_amdgcn_wmma_f32_16x16x32_bf16(false, ah, false, bh1, (short)0, acc1, false, false);
            acc1 = __builtin_amdgcn_wmma_f32_16x16x32_bf16(false, ah, false, bl1, (short)0, acc1, false, false);
            acc1 = __builtin_amdgcn_wmma_f32_16x16x32_bf16(false, al, false, bh1, (short)0, acc1, false, false);
        }
    }

    // ---- store: D frag -> M = r + 8*(lane>>4) (channel), N = lane&15 (pixel)
    const int mhi = (lane >> 4) * 8;
#pragma unroll
    for (int r = 0; r < 8; ++r) {
        const int ch   = ch0 + cw * 16 + r + mhi;
        const float bv = bias[ch];
        const int base = (b * COUT + ch) * HW + y * WIDTH;
        cc[base + n0 + m]      = acc0[r] + bv;
        cc[base + n0 + 16 + m] = acc1[r] + bv;
    }
}

// ---------------------------------------------------------------------------
// LSTM gate math + state update + leaky-ReLU output write.
// ---------------------------------------------------------------------------
__global__ __launch_bounds__(256)
void lstm_point_kernel(const float* __restrict__ cc,
                       float* __restrict__ h,
                       float* __restrict__ cst,
                       float* __restrict__ out,   // (B,T,64,H,W)
                       int tstep) {
    int idx = blockIdx.x * 256 + threadIdx.x;
    int b   = idx >> 18;
    int rem = idx & ((HID * HW) - 1);
    int ch  = rem >> 12;
    int pix = rem & (HW - 1);

    int base = (b * COUT) * HW + pix;
    float gi = cc[base + (ch + 0 * HID) * HW];
    float gf = cc[base + (ch + 1 * HID) * HW];
    float go = cc[base + (ch + 2 * HID) * HW];
    float gg = cc[base + (ch + 3 * HID) * HW];

    float i_ = 1.0f / (1.0f + __expf(-gi));
    float f_ = 1.0f / (1.0f + __expf(-gf));
    float o_ = 1.0f / (1.0f + __expf(-go));
    float g_ = tanhf(gg);

    float cn = f_ * cst[idx] + i_ * g_;
    float hn = o_ * tanhf(cn);
    cst[idx] = cn;
    h[idx]   = hn;

    float ov = (hn >= 0.0f) ? hn : NEGSL * hn;
    out[((b * TSTEPS + tstep) * HID + ch) * HW + pix] = ov;
}

// ---------------------------------------------------------------------------
// Per-channel sum / sum-of-squares over (B,T,H,W).  grid = (128, 64)
// ---------------------------------------------------------------------------
__global__ __launch_bounds__(256)
void reduce_stats_kernel(const float* __restrict__ out,
                         float* __restrict__ stats) {
    int ch = blockIdx.y;
    int bt = blockIdx.x;
    const float* base = out + (bt * HID + ch) * HW;

    float s = 0.0f, s2 = 0.0f;
    for (int i = threadIdx.x; i < HW; i += 256) {
        float v = base[i];
        s  += v;
        s2 += v * v;
    }
    __shared__ float ls[256];
    __shared__ float ls2[256];
    ls[threadIdx.x]  = s;
    ls2[threadIdx.x] = s2;
    __syncthreads();
    for (int off = 128; off > 0; off >>= 1) {
        if (threadIdx.x < off) {
            ls[threadIdx.x]  += ls[threadIdx.x + off];
            ls2[threadIdx.x] += ls2[threadIdx.x + off];
        }
        __syncthreads();
    }
    if (threadIdx.x == 0) {
        atomicAdd(&stats[ch],      ls[0]);
        atomicAdd(&stats[64 + ch], ls2[0]);
    }
}

// ---------------------------------------------------------------------------
// In-place normalization: (v - mean) * rsqrt(var + eps) * gamma + beta
// ---------------------------------------------------------------------------
__global__ __launch_bounds__(256)
void normalize_kernel(float* __restrict__ out,
                      const float* __restrict__ stats,
                      const float* __restrict__ gamma,
                      const float* __restrict__ beta) {
    int idx = blockIdx.x * 256 + threadIdx.x;
    int ch  = (idx >> 12) & (HID - 1);
    const float n = (float)(BATCH * TSTEPS * HW);
    float mean = stats[ch] / n;
    float var  = stats[64 + ch] / n - mean * mean;
    float sc   = gamma[ch] * rsqrtf(var + EPSV);
    float v = out[idx];
    out[idx] = (v - mean) * sc + beta[ch];
}

// ---------------------------------------------------------------------------
extern "C" void kernel_launch(void* const* d_in, const int* in_sizes, int n_in,
                              void* d_out, int out_size, void* d_ws, size_t ws_size,
                              hipStream_t stream) {
    const float* x     = (const float*)d_in[0];
    const float* wconv = (const float*)d_in[1];
    const float* bconv = (const float*)d_in[2];
    const float* gamma = (const float*)d_in[3];
    const float* beta  = (const float*)d_in[4];
    float* out = (float*)d_out;

    char* p = (char*)d_ws;
    float* h    = (float*)p;  p += (size_t)BATCH * HID * HW * 4;   // 8 MB
    float* c    = (float*)p;  p += (size_t)BATCH * HID * HW * 4;   // 8 MB
    float* cc   = (float*)p;  p += (size_t)BATCH * COUT * HW * 4;  // 32 MB
    float* stat = (float*)p;  p += 512;
    unsigned short* whi = (unsigned short*)p; p += (size_t)COUT * KTOT * 2;
    unsigned short* wlo = (unsigned short*)p; p += (size_t)COUT * KTOT * 2;

    init_kernel<<<(BATCH * HID * HW + 255) / 256, 256, 0, stream>>>(h, c, stat);
    prep_weights_kernel<<<(COUT * KTOT + 255) / 256, 256, 0, stream>>>(wconv, whi, wlo);

    for (int t = 0; t < TSTEPS; ++t) {
        conv_step_kernel<<<dim3(512, 4), 256, 0, stream>>>(x, h, whi, wlo, bconv, cc, t);
        lstm_point_kernel<<<(BATCH * HID * HW) / 256, 256, 0, stream>>>(cc, h, c, out, t);
    }

    reduce_stats_kernel<<<dim3(BATCH * TSTEPS, HID), 256, 0, stream>>>(out, stat);
    normalize_kernel<<<(BATCH * TSTEPS * HID * HW) / 256, 256, 0, stream>>>(out, stat, gamma, beta);
}
